// BC_Encoder_28784870818544
// MI455X (gfx1250) — matrix operand order repaired and checked by reading, not compiled, not run on hardware
//
#include <hip/hip_runtime.h>
#include <hip/hip_bf16.h>

typedef __attribute__((ext_vector_type(16))) _Float16 v16h;
typedef __attribute__((ext_vector_type(8)))  float    v8f;

#define H 256
#define TILE_M 64
#define NSEG 64
#define EPS 1e-5f

// ---- shared memory layout (bytes) ----
// halves: fragW2[65536] fragW3[65536] actbuf[16384]  -> 294912 B
// floats: W1t[768] b1[256] b2[256] b3[256] g1[256] be1[256] g2[256] be2[256]
//         postile[192] lnsum[128] lnsq[128] pool_sum[256] -> 13056 B @294912
// uint:   pool_max[256] pool_min[256]                    -> 2048 B @307968
// int:    segtile[64]                                    -> 256 B @310016
#define SMEM_BYTES 310272

__device__ __forceinline__ unsigned encf(float x) {
  unsigned b = __float_as_uint(x);
  return (b & 0x80000000u) ? ~b : (b | 0x80000000u);
}
__device__ __forceinline__ float decf(unsigned u) {
  unsigned b = (u & 0x80000000u) ? (u & 0x7FFFFFFFu) : ~u;
  return __uint_as_float(b);
}

// B-fragment address for weight element W[n][k] (B[k][n] = W[n][k]).
// frag(kt, nt) is 512 contiguous halves: lane*16 + e, lane=(n&15)+16*((kk>>3)&1), e=(kk&7)|((kk>>4)<<3)
__device__ __forceinline__ int bfrag_addr(int n, int k) {
  int kt = k >> 5, kk = k & 31;
  int lane = (n & 15) + 16 * ((kk >> 3) & 1);
  return ((kt << 4) + (n >> 4)) * 512 + lane * 16 + ((kk & 7) | ((kk >> 4) << 3));
}

// A-fragment address for activation element (m in [0,64), k in [0,256))
// 16-bit A 16x32 layout: lane=(m&15)+16*h, h=(kk>>3)&1, e=(kk&7)|((kk>>4)<<3)
__device__ __forceinline__ int afrag_addr(int m, int k) {
  int kt = k >> 5, kk = k & 31;
  int hh = (kk >> 3) & 1;
  int e  = (kk & 7) | ((kk >> 4) << 3);
  int lane = (m & 15) + 16 * hh;
  return (((m >> 4) << 3) + kt) * 512 + lane * 16 + e;
}

__device__ __forceinline__ void wmma_layer(const _Float16* actbuf, const _Float16* wfrag,
                                           const float* bias, int m_tile, int n_half,
                                           int lane, v8f (&acc)[8]) {
  int nl = lane & 15;
#pragma unroll
  for (int nt = 0; nt < 8; ++nt) {
    float bv = bias[((n_half * 8 + nt) << 4) + nl];
    v8f c;
#pragma unroll
    for (int r = 0; r < 8; ++r) c[r] = bv;
    acc[nt] = c;
  }
#pragma unroll
  for (int kt = 0; kt < 8; ++kt) {
    v16h a = *(const v16h*)(actbuf + (((m_tile << 3) + kt) * 512 + lane * 16));
#pragma unroll
    for (int nt = 0; nt < 8; ++nt) {
      v16h b = *(const v16h*)(wfrag + (((kt << 4) + n_half * 8 + nt) * 512 + lane * 16));
      acc[nt] = __builtin_amdgcn_wmma_f32_16x16x32_f16(
          false, a, false, b, (short)0, acc[nt], false, false);
    }
  }
}

__global__ __launch_bounds__(256, 1) void encoder_kernel(
    const float* __restrict__ positions, const float* __restrict__ W1,
    const float* __restrict__ b1, const float* __restrict__ W2,
    const float* __restrict__ b2, const float* __restrict__ W3,
    const float* __restrict__ b3, const float* __restrict__ g1,
    const float* __restrict__ be1, const float* __restrict__ g2,
    const float* __restrict__ be2, const int* __restrict__ batch_index,
    int Npts, int numTiles,
    float* __restrict__ gsum, unsigned* __restrict__ gmax,
    unsigned* __restrict__ gmin, float* __restrict__ gcount) {
  extern __shared__ char smem[];
  _Float16* fragW2 = (_Float16*)smem;
  _Float16* fragW3 = fragW2 + 65536;
  _Float16* actbuf = fragW3 + 65536;
  float* fbase   = (float*)(smem + 294912);
  float* W1t     = fbase;            // [3][256]
  float* b1s     = fbase + 768;
  float* b2s     = fbase + 1024;
  float* b3s     = fbase + 1280;
  float* g1s     = fbase + 1536;
  float* be1s    = fbase + 1792;
  float* g2s     = fbase + 2048;
  float* be2s    = fbase + 2304;
  float* postile = fbase + 2560;     // [64*3]
  float* lnsum   = fbase + 2752;     // [64][2]
  float* lnsq    = fbase + 2880;     // [64][2]
  float* pool_sum = fbase + 3008;    // [256]
  unsigned* pool_max = (unsigned*)(smem + 307968);
  unsigned* pool_min = (unsigned*)(smem + 308992);
  int* segtile = (int*)(smem + 310016);

  const int tid  = threadIdx.x;
  const int lane = tid & 31;
  const int wave = tid >> 5;
  const int m_tile = wave & 3;
  const int n_half = wave >> 2;
  const int nl = lane & 15;
  const int lh = lane >> 4;

  // ---- one-time per-block: stage weights into LDS in fragment layout ----
  for (int i = tid; i < H * H; i += 256) {
    int n = i >> 8, k = i & 255;
    int a = bfrag_addr(n, k);
    fragW2[a] = (_Float16)W2[i];
    fragW3[a] = (_Float16)W3[i];
  }
  for (int i = tid; i < 768; i += 256) W1t[(i % 3) * 256 + (i / 3)] = W1[i];
  if (tid < 256) {
    b1s[tid] = b1[tid];  b2s[tid] = b2[tid];  b3s[tid] = b3[tid];
    g1s[tid] = g1[tid];  be1s[tid] = be1[tid];
    g2s[tid] = g2[tid];  be2s[tid] = be2[tid];
  }
  __syncthreads();

  for (int tile = blockIdx.x; tile < numTiles; tile += gridDim.x) {
    const int tilebase = tile * TILE_M;

    // ---- stage positions + segment ids ----
    if (tid < TILE_M * 3) {
      int gi = tilebase * 3 + tid;
      postile[tid] = (gi < Npts * 3) ? positions[gi] : 0.f;
    }
    if (tid < TILE_M) {
      int gi = tilebase + tid;
      segtile[tid] = batch_index[gi < Npts ? gi : (Npts - 1)];
    }
    __syncthreads();

    // ---- layer 1 (K=3, VALU) + LN + ReLU -> actbuf (A-fragment f16) ----
    {
      const int p = tid >> 2;       // point within tile
      const int q = tid & 3;        // quarter of the 256 features
      float px = postile[p * 3 + 0];
      float py = postile[p * 3 + 1];
      float pz = postile[p * 3 + 2];
      float v[64];
      float s1 = 0.f, s2 = 0.f;
#pragma unroll
      for (int j = 0; j < 64; ++j) {
        int h = q * 64 + j;
        float y = b1s[h] + px * W1t[h] + py * W1t[256 + h] + pz * W1t[512 + h];
        v[j] = y;
        s1 += y;
        s2 += y * y;
      }
      s1 += __shfl_xor(s1, 1, 32);  s1 += __shfl_xor(s1, 2, 32);
      s2 += __shfl_xor(s2, 1, 32);  s2 += __shfl_xor(s2, 2, 32);
      float mu  = s1 * (1.f / 256.f);
      float var = s2 * (1.f / 256.f) - mu * mu;
      float inv = rsqrtf(var + EPS);
#pragma unroll
      for (int j = 0; j < 64; ++j) {
        int h = q * 64 + j;
        float y = (v[j] - mu) * inv * g1s[h] + be1s[h];
        y = fmaxf(y, 0.f);
        actbuf[afrag_addr(p, h)] = (_Float16)y;
      }
    }
    __syncthreads();

    // ---- layer 2: WMMA ----
    v8f acc[8];
    wmma_layer(actbuf, fragW2, b2s, m_tile, n_half, lane, acc);

    // ---- LayerNorm of layer-2 output (C layout) ----
    {
      float s[8], sq[8];
#pragma unroll
      for (int r = 0; r < 8; ++r) {
        float a = 0.f, b = 0.f;
#pragma unroll
        for (int nt = 0; nt < 8; ++nt) { float x = acc[nt][r]; a += x; b += x * x; }
#pragma unroll
        for (int m = 1; m < 16; m <<= 1) { a += __shfl_xor(a, m, 32); b += __shfl_xor(b, m, 32); }
        s[r] = a; sq[r] = b;
      }
      if (nl == 0) {
#pragma unroll
        for (int r = 0; r < 8; ++r) {
          int M = m_tile * 16 + r + 8 * lh;
          lnsum[M * 2 + n_half] = s[r];
          lnsq [M * 2 + n_half] = sq[r];
        }
      }
      __syncthreads();
      float mean[8], inv[8];
#pragma unroll
      for (int r = 0; r < 8; ++r) {
        int M = m_tile * 16 + r + 8 * lh;
        float su = lnsum[M * 2] + lnsum[M * 2 + 1];
        float s2 = lnsq [M * 2] + lnsq [M * 2 + 1];
        float mu = su * (1.f / 256.f);
        float var = s2 * (1.f / 256.f) - mu * mu;
        mean[r] = mu;
        inv[r]  = rsqrtf(var + EPS);
      }
#pragma unroll
      for (int nt = 0; nt < 8; ++nt) {
        int n = ((n_half * 8 + nt) << 4) + nl;
        float gg = g2s[n], bb = be2s[n];
#pragma unroll
        for (int r = 0; r < 8; ++r) {
          float y = (acc[nt][r] - mean[r]) * inv[r] * gg + bb;
          y = fmaxf(y, 0.f);
          int m = m_tile * 16 + r + 8 * lh;
          actbuf[afrag_addr(m, n)] = (_Float16)y;
        }
      }
    }
    __syncthreads();

    // reset pool scratch (no dependency on actbuf) then layer 3 WMMA
    if (tid < 256) {
      pool_sum[tid] = 0.f;
      pool_max[tid] = 0u;
      pool_min[tid] = 0xFFFFFFFFu;
    }
    wmma_layer(actbuf, fragW3, b3s, m_tile, n_half, lane, acc);
    __syncthreads();

    // ---- segmented pooling ----
    const int seg0 = segtile[0];
    const bool full = (tilebase + TILE_M) <= Npts;
    const bool uniform = full && (seg0 == segtile[TILE_M - 1]);

    if (uniform) {
#pragma unroll
      for (int nt = 0; nt < 8; ++nt) {
        float sv = 0.f, mx = -3.402823466e+38f, mn = 3.402823466e+38f;
#pragma unroll
        for (int r = 0; r < 8; ++r) {
          float x = acc[nt][r];
          sv += x; mx = fmaxf(mx, x); mn = fminf(mn, x);
        }
        sv += __shfl_xor(sv, 16, 32);
        mx = fmaxf(mx, __shfl_xor(mx, 16, 32));
        mn = fminf(mn, __shfl_xor(mn, 16, 32));
        if (lane < 16) {
          int f = ((n_half * 8 + nt) << 4) + nl;
          atomicAdd(&pool_sum[f], sv);
          atomicMax(&pool_max[f], encf(mx));
          atomicMin(&pool_min[f], encf(mn));
        }
      }
      __syncthreads();
      if (tid < 256) {
        int f = tid;
        atomicAdd(&gsum[seg0 * H + f], pool_sum[f]);
        atomicMax(&gmax[seg0 * H + f], pool_max[f]);
        atomicMin(&gmin[seg0 * H + f], pool_min[f]);
      }
    } else {
      // rare boundary/partial tile: per-point atomics
#pragma unroll
      for (int nt = 0; nt < 8; ++nt) {
        int f = ((n_half * 8 + nt) << 4) + nl;
#pragma unroll
        for (int r = 0; r < 8; ++r) {
          int m = m_tile * 16 + r + 8 * lh;
          if (tilebase + m < Npts) {
            int seg = segtile[m];
            float x = acc[nt][r];
            atomicAdd(&gsum[seg * H + f], x);
            atomicMax(&gmax[seg * H + f], encf(x));
            atomicMin(&gmin[seg * H + f], encf(x));
          }
        }
      }
    }
    if (tid < TILE_M && tilebase + tid < Npts)
      atomicAdd(&gcount[segtile[tid]], 1.f);
    __syncthreads();
  }
}

__global__ void init_ws(float* gsum, unsigned* gmax, unsigned* gmin, float* gcount) {
  int i = blockIdx.x * blockDim.x + threadIdx.x;
  if (i < NSEG * H) {
    gsum[i] = 0.f;
    gmax[i] = 0u;
    gmin[i] = 0xFFFFFFFFu;
  }
  if (i < NSEG) gcount[i] = 0.f;
}

__global__ void finalize_ws(const float* gsum, const unsigned* gmax,
                            const unsigned* gmin, const float* gcount,
                            float* out) {
  int i = blockIdx.x * blockDim.x + threadIdx.x;
  if (i < NSEG * H) {
    int b = i >> 8, f = i & 255;
    out[b * 768 + f]       = gsum[i] / gcount[b];
    out[b * 768 + 256 + f] = decf(gmax[i]);
    out[b * 768 + 512 + f] = decf(gmin[i]);
  }
}

extern "C" void kernel_launch(void* const* d_in, const int* in_sizes, int n_in,
                              void* d_out, int out_size, void* d_ws, size_t ws_size,
                              hipStream_t stream) {
  const float* positions = (const float*)d_in[0];
  const float* W1  = (const float*)d_in[1];
  const float* b1  = (const float*)d_in[2];
  const float* W2  = (const float*)d_in[3];
  const float* b2  = (const float*)d_in[4];
  const float* W3  = (const float*)d_in[5];
  const float* b3  = (const float*)d_in[6];
  const float* g1  = (const float*)d_in[7];
  const float* be1 = (const float*)d_in[8];
  const float* g2  = (const float*)d_in[9];
  const float* be2 = (const float*)d_in[10];
  const int* batch_index = (const int*)d_in[11];

  const int Npts = in_sizes[0] / 3;
  const int numTiles = (Npts + TILE_M - 1) / TILE_M;

  float*    gsum   = (float*)d_ws;
  unsigned* gmax   = (unsigned*)(gsum + NSEG * H);
  unsigned* gmin   = gmax + NSEG * H;
  float*    gcount = (float*)(gmin + NSEG * H);

  init_ws<<<(NSEG * H + 255) / 256, 256, 0, stream>>>(gsum, gmax, gmin, gcount);

  int grid = numTiles < 1024 ? numTiles : 1024;
  encoder_kernel<<<grid, 256, SMEM_BYTES, stream>>>(
      positions, W1, b1, W2, b2, W3, b3, g1, be1, g2, be2, batch_index,
      Npts, numTiles, gsum, gmax, gmin, gcount);

  finalize_ws<<<(NSEG * H + 255) / 256, 256, 0, stream>>>(gsum, gmax, gmin,
                                                          gcount, (float*)d_out);
}
